// MultiHeadAttention_75849122447753
// MI455X (gfx1250) — matrix operand order repaired and checked by reading, not compiled, not run on hardware
//
#include <hip/hip_runtime.h>
#include <hip/hip_bf16.h>

typedef __attribute__((ext_vector_type(16))) _Float16 v16h;
typedef __attribute__((ext_vector_type(8)))  float    v8f;
typedef __attribute__((ext_vector_type(4)))  unsigned int u32x4;
typedef __attribute__((ext_vector_type(8)))  int      i32x8;
typedef __attribute__((ext_vector_type(4)))  int      i32x4;

union AFrag { v16h v; unsigned u[8]; };

#define NEG_BIG (-1.0e30f)

// ---- CDNA5 Tensor Data Mover availability (6-arg clang-23 form) ----
#if defined(__has_builtin)
#if __has_builtin(__builtin_amdgcn_tensor_load_to_lds) && \
    __has_builtin(__builtin_amdgcn_s_wait_tensorcnt)
#define USE_TDM 1
#endif
#endif
#ifndef USE_TDM
#define USE_TDM 0
#endif

// ---- WMMA wrapper: D = A(16x32 f16) * B(32x16 f16) + C(16x16 f32) ----
__device__ inline v8f wmma16(const AFrag& a, const AFrag& b, v8f c) {
  return __builtin_amdgcn_wmma_f32_16x16x32_f16(
      /*neg_a=*/false, a.v, /*neg_b=*/false, b.v,
      /*c_mod=*/(short)0, c, /*reuse_a=*/false, /*reuse_b=*/false);
}

// A-fragment (16x32, row-major LDS tile, stride ldh halves).
// ISA layout: lanes 0-15 -> M=lane, VGPR i<4: K=2i,2i+1; i>=4: K=16+2(i-4)..;
// lanes 16-31 add +8 to K.
__device__ inline void load_a_frag(const _Float16* __restrict__ lds, int ldh,
                                   int m0, int k0, int lane, AFrag& f) {
  const int m  = m0 + (lane & 15);
  const int hl = (lane >> 4) & 1;
  const _Float16* row = lds + m * ldh + k0 + hl * 8;
#pragma unroll
  for (int i = 0; i < 4; ++i) {
    f.u[i]     = *(const unsigned*)(row + 2 * i);
    f.u[i + 4] = *(const unsigned*)(row + 16 + 2 * i);
  }
}

// B-fragment (32x16). LDS tile stored [n][k] row-major, stride ldh halves.
// ISA layout: lane&15 = N; VGPR i<4: K=2i,2i+1; i>=4: K=16+2(i-4)..; lanes 16-31 add +8 to K.
__device__ inline void load_b_frag(const _Float16* __restrict__ lds, int ldh,
                                   int n0, int k0, int lane, AFrag& f) {
  const int n  = n0 + (lane & 15);
  const int kb = k0 + ((lane >> 4) & 1) * 8;
  const _Float16* p = lds + n * ldh + kb;
#pragma unroll
  for (int i = 0; i < 4; ++i) {
    f.u[i]     = *(const unsigned*)(p + 2 * i);
    f.u[i + 4] = *(const unsigned*)(p + 16 + 2 * i);
  }
}

#if USE_TDM
// Issue one 2D TDM tile load (f16 elements) into LDS.
// Tile: tile_d1 rows x tile_d0 elements; global rows stride0 elements apart.
// LDS rows padded from 128B to 160B (pad_interval=32 dwords -> code 4,
// pad_amount=8 dwords -> code 7) => LDS row stride of 80 halves.
__device__ __attribute__((always_inline)) inline void
tdm_load_2d_f16(unsigned lds_off, const _Float16* gptr,
                unsigned tensor_d0, unsigned tensor_d1,
                unsigned long long stride0,
                unsigned tile_d0, unsigned tile_d1) {
  const unsigned long long ga = (unsigned long long)(size_t)gptr;
  u32x4 g0;
  g0[0] = 1u;                                          // count=1, user D#
  g0[1] = lds_off;                                     // lds_addr (bytes)
  g0[2] = (unsigned)(ga & 0xFFFFFFFFu);                // global_addr[31:0]
  g0[3] = (unsigned)((ga >> 32) & 0x01FFFFFFu) | (2u << 30);  // [56:32]|type=2
  i32x8 g1;
  g1[0] = (int)((1u << 16) |          // data_size = 2 bytes
                (1u << 20) |          // pad_enable
                (4u << 22) |          // pad_interval: 32 dwords
                (7u << 25));          // pad_amount: 8 dwords
  g1[1] = (int)((tensor_d0 & 0xFFFFu) << 16);
  g1[2] = (int)(((tensor_d0 >> 16) & 0xFFFFu) | ((tensor_d1 & 0xFFFFu) << 16));
  g1[3] = (int)(((tensor_d1 >> 16) & 0xFFFFu) | (tile_d0 << 16));
  g1[4] = (int)(tile_d1 & 0xFFFFu);                    // tile_dim1, tile_dim2=0
  g1[5] = (int)(unsigned)(stride0 & 0xFFFFFFFFull);    // dim0 stride lo
  g1[6] = (int)(unsigned)((stride0 >> 32) & 0xFFFFull);
  g1[7] = 0;
  i32x4 zz4; zz4[0] = 0; zz4[1] = 0; zz4[2] = 0; zz4[3] = 0;
  i32x8 zz8;
#pragma unroll
  for (int i = 0; i < 8; ++i) zz8[i] = 0;
  __builtin_amdgcn_tensor_load_to_lds(g0, g1, zz4, zz4, zz8, 0);
}
#endif

// =====================================================================
// Kernel 1: fused QKV projection.  Yz = x @ Wz + bz  (z = Q,K,V)
//   Q -> f16 [B,H,T,D], pre-scaled by 1/sqrt(D)=0.125
//   K,V -> f16 transposed [B,H,D,T]
// Tile: 128x128 per 256-thread block, K-step 32.
// =====================================================================
__global__ __launch_bounds__(256)
void qkv_kernel(const float* __restrict__ x,
                const float* __restrict__ Wq, const float* __restrict__ bq,
                const float* __restrict__ Wk, const float* __restrict__ bk,
                const float* __restrict__ Wv, const float* __restrict__ bv,
                _Float16* __restrict__ qout, _Float16* __restrict__ ktout,
                _Float16* __restrict__ vtout) {
  const int z = blockIdx.z;
  const float* W    = (z == 0) ? Wq : (z == 1) ? Wk : Wv;
  const float* bias = (z == 0) ? bq : (z == 1) ? bk : bv;

  __shared__ __align__(16) _Float16 Alds[128 * 40];
  __shared__ __align__(16) _Float16 Blds[128 * 40];  // transposed [n][k]

  const int tid  = threadIdx.x;
  const int lane = tid & 31;
  const int wid  = tid >> 5;
  const int wm   = (wid & 3) * 32;   // 4 wave-rows of 32
  const int wn   = (wid >> 2) * 64;  // 2 wave-cols of 64

  const int m_base = blockIdx.x * 128;
  const int n_base = blockIdx.y * 128;

  v8f acc[2][4];
#pragma unroll
  for (int mt = 0; mt < 2; ++mt)
#pragma unroll
    for (int nt = 0; nt < 4; ++nt)
#pragma unroll
      for (int r = 0; r < 8; ++r) acc[mt][nt][r] = 0.0f;

  for (int k0 = 0; k0 < 1024; k0 += 32) {
    __syncthreads();
    {  // stage A: 128x32 fp32 -> f16, row-major [m][k]
      const int row = tid >> 1;
      const int col = (tid & 1) * 16;
      const float* src = x + (size_t)(m_base + row) * 1024 + k0 + col;
      _Float16* dst = Alds + row * 40 + col;
#pragma unroll
      for (int i = 0; i < 16; ++i) dst[i] = (_Float16)src[i];
      if (k0 + 32 < 1024) __builtin_prefetch(src + 32, 0, 1);
    }
    {  // stage B transposed: W[k][n] -> Blds[n][k]
      const int k  = tid >> 3;
      const int nc = (tid & 7) * 16;
      const float* src = W + (size_t)(k0 + k) * 1024 + n_base + nc;
#pragma unroll
      for (int i = 0; i < 16; ++i) Blds[(nc + i) * 40 + k] = (_Float16)src[i];
      if (k0 + 32 < 1024) __builtin_prefetch(src + 32 * 1024, 0, 1);
    }
    __syncthreads();

    AFrag a0, a1;
    load_a_frag(Alds, 40, wm, 0, lane, a0);
    load_a_frag(Alds, 40, wm + 16, 0, lane, a1);
#pragma unroll
    for (int nt = 0; nt < 4; ++nt) {
      AFrag b;
      load_b_frag(Blds, 40, wn + nt * 16, 0, lane, b);
      acc[0][nt] = wmma16(a0, b, acc[0][nt]);
      acc[1][nt] = wmma16(a1, b, acc[1][nt]);
    }
  }

  // epilogue: C layout -> global (with layout transform per z)
  const int hl = lane >> 4;
  const int nn = lane & 15;
#pragma unroll
  for (int mt = 0; mt < 2; ++mt) {
#pragma unroll
    for (int nt = 0; nt < 4; ++nt) {
      const int cbase = n_base + wn + nt * 16 + nn;
      const float bv_ = bias[cbase];
#pragma unroll
      for (int r = 0; r < 8; ++r) {
        const int mrow = m_base + wm + mt * 16 + r + hl * 8;
        const float val = acc[mt][nt][r] + bv_;
        const int b = mrow >> 11, t = mrow & 2047;
        const int h = cbase >> 6, d = cbase & 63;
        if (z == 0) {
          qout[(((size_t)(b * 16 + h)) * 2048 + t) * 64 + d] =
              (_Float16)(val * 0.125f);
        } else {
          _Float16* o = (z == 1) ? ktout : vtout;
          o[(((size_t)(b * 16 + h)) * 64 + d) * 2048 + t] = (_Float16)val;
        }
      }
    }
  }
}

// =====================================================================
// Kernel 2: causal flash attention.
// grid = (B*H, T/64); block = 128 threads (4 waves), wave w owns rows
// [i0+16w, i0+16w+15] with register-resident online softmax.
// Q/V tiles staged by the Tensor Data Mover when available.
// =====================================================================
__global__ __launch_bounds__(128)
void attn_kernel(const _Float16* __restrict__ q, const _Float16* __restrict__ kt,
                 const _Float16* __restrict__ vt, _Float16* __restrict__ o) {
  __shared__ __align__(16) _Float16 Qlds[64 * 80];       // [t][d]
  __shared__ __align__(16) _Float16 Klds[64 * 80];       // [s][d]
  __shared__ __align__(16) _Float16 Vlds[64 * 80];       // [d][s]
  __shared__ __align__(16) _Float16 Plds[4 * 16 * 80];   // per-wave [m][s]

  const int bh = blockIdx.x;          // b*16+h
  const int i0 = blockIdx.y * 64;     // query row block
  const int tid = threadIdx.x, lane = tid & 31, w = tid >> 5;
  const int hl = lane >> 4, nn = lane & 15;

  // ---- stage Q (64x64) once ----
#if USE_TDM
  if (tid < 32) {
    tdm_load_2d_f16((unsigned)(size_t)(&Qlds[0]),
                    q + (((size_t)bh * 2048) + i0) * 64,
                    /*tensor_d0=*/64, /*tensor_d1=*/2048, /*stride0=*/64,
                    /*tile_d0=*/64, /*tile_d1=*/64);
    __builtin_amdgcn_s_wait_tensorcnt(0);
  }
#else
  {
    const int row = tid >> 1, col = (tid & 1) * 32;
    const _Float16* src = q + (((size_t)bh * 2048) + i0 + row) * 64 + col;
    _Float16* dst = Qlds + row * 80 + col;
#pragma unroll
    for (int i = 0; i < 4; ++i) ((uint4*)dst)[i] = ((const uint4*)src)[i];
  }
#endif
  __syncthreads();
  AFrag qa0, qa1;
  load_a_frag(Qlds, 80, w * 16, 0, lane, qa0);
  load_a_frag(Qlds, 80, w * 16, 32, lane, qa1);

  v8f oacc[4];
  float mrun[8], lrun[8];
#pragma unroll
  for (int dt = 0; dt < 4; ++dt)
#pragma unroll
    for (int r = 0; r < 8; ++r) oacc[dt][r] = 0.0f;
#pragma unroll
  for (int r = 0; r < 8; ++r) { mrun[r] = NEG_BIG; lrun[r] = 0.0f; }

  const int trow_base = i0 + w * 16 + hl * 8;
  const int nkb = (i0 >> 6) + 1;

  for (int kb = 0; kb < nkb; ++kb) {
    const int s0 = kb * 64;
    __syncthreads();
#if USE_TDM
    if (tid < 32) {  // V tile [d][s] direct 2D copy via TDM (overlaps K stage)
      tdm_load_2d_f16((unsigned)(size_t)(&Vlds[0]),
                      vt + ((size_t)bh * 64) * 2048 + s0,
                      /*tensor_d0=*/2048, /*tensor_d1=*/64, /*stride0=*/2048,
                      /*tile_d0=*/64, /*tile_d1=*/64);
    }
#endif
    {  // stage K (transpose to [s][d]) with all 128 threads
      const int d = tid >> 1, sc = (tid & 1) * 32;
      const _Float16* ks = kt + ((size_t)bh * 64 + d) * 2048 + s0 + sc;
      union { uint4 q4[4]; _Float16 h[32]; } tmp;
#pragma unroll
      for (int i = 0; i < 4; ++i) tmp.q4[i] = ((const uint4*)ks)[i];
#pragma unroll
      for (int i = 0; i < 32; ++i) Klds[(sc + i) * 80 + d] = tmp.h[i];
#if !USE_TDM
      const _Float16* vs = vt + ((size_t)bh * 64 + d) * 2048 + s0 + sc;
      _Float16* vd = Vlds + d * 80 + sc;
#pragma unroll
      for (int i = 0; i < 4; ++i) ((uint4*)vd)[i] = ((const uint4*)vs)[i];
#endif
    }
#if USE_TDM
    if (tid < 32) __builtin_amdgcn_s_wait_tensorcnt(0);
#endif
    __syncthreads();

    // ---- S = Q K^T (scale pre-folded into Q) ----
    v8f sacc[4];
#pragma unroll
    for (int nt = 0; nt < 4; ++nt) {
#pragma unroll
      for (int r = 0; r < 8; ++r) sacc[nt][r] = 0.0f;
      AFrag b0, b1;
      load_b_frag(Klds, 80, nt * 16, 0, lane, b0);
      load_b_frag(Klds, 80, nt * 16, 32, lane, b1);
      sacc[nt] = wmma16(qa0, b0, sacc[nt]);
      sacc[nt] = wmma16(qa1, b1, sacc[nt]);
    }

    // ---- causal mask (only the diagonal block can violate causality) ----
    if (s0 + 63 > i0 + w * 16) {
#pragma unroll
      for (int nt = 0; nt < 4; ++nt) {
        const int s = s0 + nt * 16 + nn;
#pragma unroll
        for (int r = 0; r < 8; ++r)
          if (s > trow_base + r) sacc[nt][r] = NEG_BIG;
      }
    }

    // ---- online softmax (per 16-lane half; each half owns 8 rows) ----
    float pv[4][8];
#pragma unroll
    for (int r = 0; r < 8; ++r) {
      float v = fmaxf(fmaxf(sacc[0][r], sacc[1][r]),
                      fmaxf(sacc[2][r], sacc[3][r]));
      v = fmaxf(v, __shfl_xor(v, 1));
      v = fmaxf(v, __shfl_xor(v, 2));
      v = fmaxf(v, __shfl_xor(v, 4));
      v = fmaxf(v, __shfl_xor(v, 8));
      const float nm = fmaxf(mrun[r], v);
      const float alpha = __expf(mrun[r] - nm);
      mrun[r] = nm;
      float rs = 0.0f;
#pragma unroll
      for (int nt = 0; nt < 4; ++nt) {
        const float p = __expf(sacc[nt][r] - nm);
        pv[nt][r] = p;
        rs += p;
      }
      rs += __shfl_xor(rs, 1);
      rs += __shfl_xor(rs, 2);
      rs += __shfl_xor(rs, 4);
      rs += __shfl_xor(rs, 8);
      lrun[r] = lrun[r] * alpha + rs;
#pragma unroll
      for (int dt = 0; dt < 4; ++dt) oacc[dt][r] *= alpha;
    }

    // ---- P -> LDS (C layout -> row-major [m][s]) -> A-fragments ----
    _Float16* pw = Plds + w * 16 * 80;
#pragma unroll
    for (int nt = 0; nt < 4; ++nt)
#pragma unroll
      for (int r = 0; r < 8; ++r)
        pw[(r + hl * 8) * 80 + nt * 16 + nn] = (_Float16)pv[nt][r];

    AFrag pa0, pa1;
    load_a_frag(pw, 80, 0, 0, lane, pa0);
    load_a_frag(pw, 80, 0, 32, lane, pa1);

    // ---- O += P V ----
#pragma unroll
    for (int dt = 0; dt < 4; ++dt) {
      AFrag vb0, vb1;
      load_b_frag(Vlds, 80, dt * 16, 0, lane, vb0);
      load_b_frag(Vlds, 80, dt * 16, 32, lane, vb1);
      oacc[dt] = wmma16(pa0, vb0, oacc[dt]);
      oacc[dt] = wmma16(pa1, vb1, oacc[dt]);
    }
  }

  // ---- epilogue: O/l -> f16 [B][T][H*D] ----
  const int b = bh >> 4, h = bh & 15;
#pragma unroll
  for (int r = 0; r < 8; ++r) {
    const int t = trow_base + r;
    const float inv = 1.0f / lrun[r];
#pragma unroll
    for (int dt = 0; dt < 4; ++dt)
      o[((size_t)b * 2048 + t) * 1024 + h * 64 + dt * 16 + nn] =
          (_Float16)(oacc[dt][r] * inv);
  }
}

// =====================================================================
// Kernel 3: output projection.  out = O @ Wp + bp  (fp32 out)
// =====================================================================
__global__ __launch_bounds__(256)
void proj_kernel(const _Float16* __restrict__ O, const float* __restrict__ Wp,
                 const float* __restrict__ bp, float* __restrict__ out) {
  __shared__ __align__(16) _Float16 Alds[128 * 40];
  __shared__ __align__(16) _Float16 Blds[128 * 40];

  const int tid = threadIdx.x, lane = tid & 31, wid = tid >> 5;
  const int wm = (wid & 3) * 32;
  const int wn = (wid >> 2) * 64;
  const int m_base = blockIdx.x * 128;
  const int n_base = blockIdx.y * 128;

  v8f acc[2][4];
#pragma unroll
  for (int mt = 0; mt < 2; ++mt)
#pragma unroll
    for (int nt = 0; nt < 4; ++nt)
#pragma unroll
      for (int r = 0; r < 8; ++r) acc[mt][nt][r] = 0.0f;

  for (int k0 = 0; k0 < 1024; k0 += 32) {
    __syncthreads();
    {  // stage A (already f16): vector copy
      const int row = tid >> 1;
      const int col = (tid & 1) * 16;
      const _Float16* src = O + (size_t)(m_base + row) * 1024 + k0 + col;
      _Float16* dst = Alds + row * 40 + col;
      ((uint4*)dst)[0] = ((const uint4*)src)[0];
      ((uint4*)dst)[1] = ((const uint4*)src)[1];
      if (k0 + 32 < 1024) __builtin_prefetch(src + 32, 0, 1);
    }
    {  // stage B transposed
      const int k  = tid >> 3;
      const int nc = (tid & 7) * 16;
      const float* src = Wp + (size_t)(k0 + k) * 1024 + n_base + nc;
#pragma unroll
      for (int i = 0; i < 16; ++i) Blds[(nc + i) * 40 + k] = (_Float16)src[i];
      if (k0 + 32 < 1024) __builtin_prefetch(src + 32 * 1024, 0, 1);
    }
    __syncthreads();

    AFrag a0, a1;
    load_a_frag(Alds, 40, wm, 0, lane, a0);
    load_a_frag(Alds, 40, wm + 16, 0, lane, a1);
#pragma unroll
    for (int nt = 0; nt < 4; ++nt) {
      AFrag b;
      load_b_frag(Blds, 40, wn + nt * 16, 0, lane, b);
      acc[0][nt] = wmma16(a0, b, acc[0][nt]);
      acc[1][nt] = wmma16(a1, b, acc[1][nt]);
    }
  }

  const int hl = lane >> 4, nn = lane & 15;
#pragma unroll
  for (int mt = 0; mt < 2; ++mt) {
#pragma unroll
    for (int nt = 0; nt < 4; ++nt) {
      const int c = n_base + wn + nt * 16 + nn;
      const float bv_ = bp[c];
#pragma unroll
      for (int r = 0; r < 8; ++r) {
        const int mrow = m_base + wm + mt * 16 + r + hl * 8;
        out[(size_t)mrow * 1024 + c] = acc[mt][nt][r] + bv_;
      }
    }
  }
}

// =====================================================================
extern "C" void kernel_launch(void* const* d_in, const int* in_sizes, int n_in,
                              void* d_out, int out_size, void* d_ws,
                              size_t ws_size, hipStream_t stream) {
  const float* x  = (const float*)d_in[0];
  const float* Wq = (const float*)d_in[1];
  const float* bq = (const float*)d_in[2];
  const float* Wk = (const float*)d_in[3];
  const float* bk = (const float*)d_in[4];
  const float* Wv = (const float*)d_in[5];
  const float* bv = (const float*)d_in[6];
  const float* Wp = (const float*)d_in[7];
  const float* bp = (const float*)d_in[8];
  float* out = (float*)d_out;

  // workspace: 4 x 16 MB f16 tensors (Q, Kt, Vt, O)
  char* ws = (char*)d_ws;
  const size_t SEG = (size_t)8192 * 1024 * 2;  // 16 MB
  _Float16* q  = (_Float16*)(ws + 0 * SEG);
  _Float16* kt = (_Float16*)(ws + 1 * SEG);
  _Float16* vt = (_Float16*)(ws + 2 * SEG);
  _Float16* o  = (_Float16*)(ws + 3 * SEG);

  dim3 g1(64, 8, 3);  // (8192/128, 1024/128, {Q,K,V})
  qkv_kernel<<<g1, 256, 0, stream>>>(x, Wq, bq, Wk, bk, Wv, bv, q, kt, vt);

  dim3 g2(64, 32);    // (B*H, T/64)
  attn_kernel<<<g2, 128, 0, stream>>>(q, kt, vt, o);

  dim3 g3(64, 8);     // (8192/128, 1024/128)
  proj_kernel<<<g3, 256, 0, stream>>>(o, Wp, bp, out);
}
